// DAM_41472204210255
// MI455X (gfx1250) — matrix operand order repaired and checked by reading, not compiled, not run on hardware
//
#include <hip/hip_runtime.h>

// Problem constants (reference: B=2, C=32, N=4096)
#define BB 2
#define CC 32
#define NN 4096
#define CN (CC * NN)   // 131072
#define BNE (BB * NN)  // 8192 elements per channel for BN stats

typedef _Float16 v16h __attribute__((ext_vector_type(16)));
typedef _Float16 v8h  __attribute__((ext_vector_type(8)));
typedef float    v8f  __attribute__((ext_vector_type(8)));
typedef float    v4f  __attribute__((ext_vector_type(4)));

struct HPair { v8h a, b; };

static __device__ __forceinline__ v16h ld16(const _Float16* p0, const _Float16* p1) {
  HPair t;
  t.a = *(const v8h*)p0;   // global_load_b128 / ds_load_b128
  t.b = *(const v8h*)p1;
  return __builtin_bit_cast(v16h, t);
}

static __device__ __forceinline__ v8f wmma16(v16h a, v16h b, v8f c) {
  // D = A(16x32 f16) * B(32x16 f16) + C(16x16 f32)
  return __builtin_amdgcn_wmma_f32_16x16x32_f16(false, a, false, b, (short)0, c,
                                                false, false);
}

// Raw hardware transcendentals (v_exp_f32 / v_log_f32).  Safe here: exp2
// arguments are bounded above (~<=80) and anything below -126 must produce a
// ~0 softmax weight, which matches v_exp_f32's flush-to-zero behavior.
static __device__ __forceinline__ float fexp2(float x) { return __builtin_amdgcn_exp2f(x); }
static __device__ __forceinline__ float flog2(float x) { return __builtin_amdgcn_logf(x); }

// ---------------------------------------------------------------------------
// Kernel 1: BatchNorm1d (training stats over (B,N) per channel) + ReLU.
// Writes f16 copies in two layouts:
//   f_t  [b][n][c]  (rows of 32 f16 = 64B; feeds score-WMMA A and B tiles)
//   f_ct [b][c][n]  (feeds projection-WMMA A tiles, K = n)
// ---------------------------------------------------------------------------
__global__ void bn_relu_kernel(const float* __restrict__ F,
                               const float* __restrict__ gamma,
                               const float* __restrict__ bias,
                               _Float16* __restrict__ f_t,
                               _Float16* __restrict__ f_ct) {
  const int c = blockIdx.x;       // 32 channels
  const int t = threadIdx.x;      // 256 threads
  float s = 0.f, sq = 0.f;
  for (int i = 0; i < BNE / 256; ++i) {
    int j = t + i * 256;
    int b = j >> 12, n = j & (NN - 1);
    float v = F[b * CN + c * NN + n];
    s += v; sq += v * v;
  }
  __shared__ float r1[256], r2[256];
  r1[t] = s; r2[t] = sq;
  __syncthreads();
  for (int off = 128; off > 0; off >>= 1) {
    if (t < off) { r1[t] += r1[t + off]; r2[t] += r2[t + off]; }
    __syncthreads();
  }
  float mean = r1[0] * (1.f / BNE);
  float var  = r2[0] * (1.f / BNE) - mean * mean;   // biased variance
  float scale = rsqrtf(var + 1e-5f) * gamma[c];
  float shift = bias[c] - mean * scale;
  for (int i = 0; i < BNE / 256; ++i) {
    int j = t + i * 256;
    int b = j >> 12, n = j & (NN - 1);
    float v = F[b * CN + c * NN + n];
    float fv = fmaxf(v * scale + shift, 0.f);
    _Float16 h = (_Float16)fv;
    f_t[(b * NN + n) * CC + c]  = h;
    f_ct[(b * CC + c) * NN + n] = h;
  }
}

// ---------------------------------------------------------------------------
// Kernel 2: aa[b][d][c] = sum_n a[b,c,n]*a[b,d,n]   (einsum 'bcn,bdn->bdc')
// One block per (b,d): 256 threads = 32 channels x 8 n-slices.
// ---------------------------------------------------------------------------
__global__ void aa_kernel(const float* __restrict__ A, float* __restrict__ aa) {
  const int bd = blockIdx.x;            // b*32 + d
  const int b = bd >> 5, d = bd & 31;
  const int t = threadIdx.x;
  const int c = t & 31, sl = t >> 5;    // slice 0..7
  const float* ad = A + b * CN + d * NN;
  const float* ac = A + b * CN + c * NN;
  float p = 0.f;
  for (int n = sl * 512; n < sl * 512 + 512; ++n) p += ac[n] * ad[n];
  __shared__ float red[256];
  red[t] = p;
  __syncthreads();
  for (int off = 128; off >= 32; off >>= 1) {
    if (t < off) red[t] += red[t + off];
    __syncthreads();
  }
  if (t < 32) aa[(b * 32 + d) * 32 + t] = red[t];
}

// ---------------------------------------------------------------------------
// Kernel 3: x = softmax(aa, axis=0) over the batch (2 entries, max-subtracted:
// logits here can reach +-few hundred, so stabilization is mandatory).
// ---------------------------------------------------------------------------
__global__ void x_kernel(const float* __restrict__ aa, float* __restrict__ x) {
  int g = blockIdx.x * 256 + threadIdx.x;   // 0..1023 = (j,i) pairs
  float a0 = aa[g], a1 = aa[1024 + g];
  float m = fmaxf(a0, a1);
  float e0 = fexp2(1.44269504f * (a0 - m));
  float e1 = fexp2(1.44269504f * (a1 - m));
  float inv = 1.f / (e0 + e1);
  x[g] = e0 * inv;
  x[1024 + g] = e1 * inv;
}

// ---------------------------------------------------------------------------
// Kernel 4: base output = 2*a + beta * e_c, with
//   e_c[b,c,n] = sum_j x[b,j,c] * a[b,j,n]
// Writes every element of d_out (clears the harness poison).
// ---------------------------------------------------------------------------
__global__ void cam_kernel(const float* __restrict__ A, const float* __restrict__ x,
                           const float* __restrict__ betaP, float* __restrict__ out) {
  int idx = blockIdx.x * 256 + threadIdx.x;   // 0..262143
  int b = idx / CN;
  int rem = idx - b * CN;
  int c = rem >> 12, n = rem & (NN - 1);
  const float* xb = x + b * 1024;
  const float* ab = A + b * CN;
  float acc = 0.f;
#pragma unroll
  for (int j = 0; j < 32; ++j) acc += xb[j * 32 + c] * ab[j * NN + n];
  out[idx] = 2.f * A[idx] + betaP[0] * acc;
}

// ---------------------------------------------------------------------------
// Kernel 5 (PAM pass 1): L2D[b,n] = log2( sum_p exp(<f_p, f_n>) ).
// One wave per 16-column n-tile; 256 score WMMAs per wave (K = C = 32).
// The B fragment is pre-scaled by log2(e) so the WMMA emits scores directly
// in the exp2 domain.  Accumulation uses 8 independent slots (one per
// C-layout row register) so the v_add chain never serializes behind a
// single register -- only an 8-way tree reduction at the end.
// ---------------------------------------------------------------------------
__global__ void __launch_bounds__(32) pam_pass1(const _Float16* __restrict__ f_t,
                                                float* __restrict__ L2D) {
  const int blk = blockIdx.x;
  const int b = blk >> 8, nt = blk & 255;
  const int lane = threadIdx.x;
  const int lo = lane & 15, hi = lane >> 4;
  const _Float16* ft = f_t + b * (NN * CC);
  // B fragment: column n = nt*16+lo, K(channel) in {hi*8..+8} u {hi*8+16..+8}
  const _Float16* pb = ft + (nt * 16 + lo) * CC + hi * 8;
  v16h bf = ld16(pb, pb + 16) * (_Float16)1.44269504f;   // exp2-domain prescale
  v8f accv = {};                                        // 8 independent sums
#pragma unroll 2
  for (int pt = 0; pt < 256; ++pt) {
    // A fragment: row p = pt*16+lo, K in {hi*8..+8} u {hi*8+16..+8}
    const _Float16* pa = ft + (pt * 16 + lo) * CC + hi * 8;
    v16h af = ld16(pa, pa + 16);
    v8f z = {};
    v8f s = wmma16(af, bf, z);   // log2e * scores[p_local][n_local]
#pragma unroll
    for (int r = 0; r < 8; ++r) accv[r] += fexp2(s[r]);
  }
  // tree-reduce the 8 slots, then combine complementary row-halves
  float a01 = accv[0] + accv[1], a23 = accv[2] + accv[3];
  float a45 = accv[4] + accv[5], a67 = accv[6] + accv[7];
  float acc = (a01 + a23) + (a45 + a67);
  float tot = acc + __shfl_xor(acc, 16, 32);
  if (lane < 16) L2D[b * NN + nt * 16 + lane] = flog2(tot);
}

// ---------------------------------------------------------------------------
// Kernel 6 (PAM pass 2, fused flash-style): for each 16-wide m-tile,
// loop n-chunks of 32: recompute score tiles with WMMA (B pre-scaled by
// log2e), form w[n,m] = exp2(score2 - L2D[n]) in f16 (always in (0,1]),
// transpose through 1KB LDS into B-fragment layout, and feed the
// projection WMMAs acc[c, m] += f[c, n-chunk] * w.  Epilogue adds
// alpha*e_p into the CAM base already residing in d_out.
// ---------------------------------------------------------------------------
__global__ void __launch_bounds__(32) pam_pass2(const _Float16* __restrict__ f_t,
                                                const _Float16* __restrict__ f_ct,
                                                const float* __restrict__ L2D,
                                                const float* __restrict__ alphaP,
                                                float* __restrict__ out) {
  __shared__ v8h wlds8[64];                 // 16 rows (m) x 32 cols (n) f16
  _Float16* wlds = (_Float16*)wlds8;
  const int blk = blockIdx.x;
  const int b = blk >> 8, mt = blk & 255;
  const int lane = threadIdx.x;
  const int lo = lane & 15, hi = lane >> 4;
  const _Float16* ft = f_t + b * (NN * CC);
  // Loop-invariant B fragment for the score WMMAs: columns = m-tile,
  // pre-scaled by log2(e) -> scores come out in the exp2 domain.
  const _Float16* pbm = ft + (mt * 16 + lo) * CC + hi * 8;
  v16h bm = ld16(pbm, pbm + 16) * (_Float16)1.44269504f;
  v8f acc0 = {}, acc1 = {};
  const float* l2b = L2D + b * NN;
  const _Float16* fc0 = f_ct + (b * CC + lo) * NN;        // channels 0..15
  const _Float16* fc1 = f_ct + (b * CC + 16 + lo) * NN;   // channels 16..31
  for (int it = 0; it < 128; ++it) {
    const int n0 = it * 32;
    const _Float16* pa0 = ft + (n0 + lo) * CC + hi * 8;        // n rows 0..15
    const _Float16* pa1 = ft + (n0 + 16 + lo) * CC + hi * 8;   // n rows 16..31
    v16h a0 = ld16(pa0, pa0 + 16);
    v16h a1 = ld16(pa1, pa1 + 16);
    v8f z = {};
    v8f s0 = wmma16(a0, bm, z);   // log2e*scores[n_local][m_local], n_local<16
    v8f s1 = wmma16(a1, bm, z);   // n_local in 16..31
    // L2D for this lane's score rows: n = n0 + hi*8 + r (+16 for s1)
    const float* lp = l2b + n0 + hi * 8;
    v4f q0 = *(const v4f*)(lp);
    v4f q1 = *(const v4f*)(lp + 4);
    v4f q2 = *(const v4f*)(lp + 16);
    v4f q3 = *(const v4f*)(lp + 20);
    v8h h0, h1;
#pragma unroll
    for (int r = 0; r < 8; ++r) {
      float l2a = (r < 4) ? q0[r] : q1[r - 4];
      float l2c = (r < 4) ? q2[r] : q3[r - 4];
      h0[r] = (_Float16)fexp2(s0[r] - l2a);
      h1[r] = (_Float16)fexp2(s1[r] - l2c);
    }
    __syncthreads();   // single-wave WG: cheap; orders LDS reuse
    // C-layout -> LDS [m][n]: row m = lo, cols hi*8..+8 (s0) / +16 (s1)
    *(v8h*)&wlds[lo * 32 + hi * 8] = h0;
    *(v8h*)&wlds[lo * 32 + 16 + hi * 8] = h1;
    __syncthreads();
    // B fragment of w: column m = lo, K(n) in {hi*8..+8} u {hi*8+16..+8}
    const _Float16* pw = &wlds[lo * 32 + hi * 8];
    v16h bw = ld16(pw, pw + 16);
    // A fragments: rows = channels, K = n-chunk
    const _Float16* pc0 = fc0 + n0 + hi * 8;
    const _Float16* pc1 = fc1 + n0 + hi * 8;
    v16h c0 = ld16(pc0, pc0 + 16);
    v16h c1 = ld16(pc1, pc1 + 16);
    acc0 = wmma16(c0, bw, acc0);
    acc1 = wmma16(c1, bw, acc1);
  }
  const float alpha = alphaP[0];
  const int m = mt * 16 + lo;
#pragma unroll
  for (int r = 0; r < 8; ++r) {
    int c = hi * 8 + r;
    int i0 = b * CN + c * NN + m;
    out[i0] = out[i0] + alpha * acc0[r];
    int i1 = b * CN + (c + 16) * NN + m;
    out[i1] = out[i1] + alpha * acc1[r];
  }
}

// ---------------------------------------------------------------------------
extern "C" void kernel_launch(void* const* d_in, const int* in_sizes, int n_in,
                              void* d_out, int out_size, void* d_ws, size_t ws_size,
                              hipStream_t stream) {
  const float* F     = (const float*)d_in[0];   // Feature [2,32,4096]
  const float* gamma = (const float*)d_in[1];   // [32]
  const float* bias  = (const float*)d_in[2];   // [32]
  const float* alpha = (const float*)d_in[3];   // scalar
  const float* beta  = (const float*)d_in[4];   // scalar
  float* out = (float*)d_out;                   // [2,32,4096] f32

  char* ws = (char*)d_ws;                       // ~1.07 MB used
  _Float16* f_t  = (_Float16*)(ws);             // [b][n][c] f16, 524288 B
  _Float16* f_ct = (_Float16*)(ws + 524288);    // [b][c][n] f16, 524288 B
  float*    L2D  = (float*)(ws + 1048576);      // [b][n] f32,    32768 B
  float*    aaB  = (float*)(ws + 1081344);      // [2][32][32],    8192 B
  float*    xB   = (float*)(ws + 1089536);      // [2][32][32],    8192 B

  bn_relu_kernel<<<32, 256, 0, stream>>>(F, gamma, bias, f_t, f_ct);
  aa_kernel<<<64, 256, 0, stream>>>(F, aaB);
  x_kernel<<<4, 256, 0, stream>>>(aaB, xB);
  cam_kernel<<<1024, 256, 0, stream>>>(F, xB, beta, out);     // writes all of d_out
  pam_pass1<<<BB * 256, 32, 0, stream>>>(f_t, L2D);
  pam_pass2<<<BB * 256, 32, 0, stream>>>(f_t, f_ct, L2D, alpha, out);
}
